// MipNerf360Field_77094662963294
// MI455X (gfx1250) — matrix operand order repaired and checked by reading, not compiled, not run on hardware
//
#include <hip/hip_runtime.h>

typedef __bf16 v16bf __attribute__((ext_vector_type(16)));
typedef float  v8f   __attribute__((ext_vector_type(8)));
typedef unsigned int v4u __attribute__((ext_vector_type(4)));

union Frag { v16bf v; v4u u[2]; };

#define NB        21          // icosahedron basis vectors
#define NDEG      12
#define FEAT      504         // 2 * NDEG * NB
#define FEAT_PAD  512
#define LDF       520         // feat LDS row stride (elements) -> 1040B, 4-bank skew/row
#define LDA       264         // act  LDS row stride (elements) ->  528B, 4-bank skew/row
#define MT        128         // samples per workgroup
#define NW        256         // net width

// bf16 workspace element offsets (pre-swizzled weights)
#define OFF_W0   0            // 256*512
#define OFF_W14  131072       // 4 * 256*256  (layers 1..4)
#define OFF_W5A  393216       // 256*256      (skip layer, x part)
#define OFF_W5B  458752       // 256*512      (skip layer, feat part, padded)
#define OFF_W6   589824       // 256*256
#define OFF_W7   655360       // 256*256

struct Biases { const float* b[8]; };

__device__ __forceinline__ __bf16 f2bf(float f) {
  unsigned u = __builtin_bit_cast(unsigned, f);
  u += 0x7FFFu + ((u >> 16) & 1u);            // round-to-nearest-even
  unsigned short h = (unsigned short)(u >> 16);
  return __builtin_bit_cast(__bf16, h);
}
__device__ __forceinline__ float bf2f(__bf16 b) {
  unsigned short h = __builtin_bit_cast(unsigned short, b);
  unsigned u = (unsigned)h << 16;
  return __builtin_bit_cast(float, u);
}

// ---------------------------------------------------------------------------
// Weight prep: fp32 (Ktot x 256, row-major) -> bf16, padded to Kpad, laid out
// in WMMA B-fragment order: block(nt,ks) of 512 elems, lane = (khalf<<4)|nl,
// lane holds 16 contiguous K values of column n = nt*16+nl.
// ---------------------------------------------------------------------------
__global__ void prep_wt(const float* __restrict__ src, __bf16* __restrict__ dst,
                        int Ksrc, int Kpad, int rowOff) {
  int e = blockIdx.x * blockDim.x + threadIdx.x;
  int total = NW * Kpad;
  if (e >= total) return;
  int kk      = e & 15;
  int lane    = (e >> 4) & 31;
  int blk     = e >> 9;
  int kblocks = Kpad >> 5;
  int ks = blk % kblocks;
  int nt = blk / kblocks;
  int n  = (nt << 4) + (lane & 15);
  int k  = (ks << 5) + ((lane >> 4) << 4) + kk;
  float v = (k < Ksrc) ? src[(size_t)(rowOff + k) * NW + n] : 0.0f;
  dst[e] = f2bf(v);
}

// ---------------------------------------------------------------------------
// Per-wave GEMM accumulate: 128 rows (8 M-tiles) x 32 cols (2 N-tiles owned
// by this wave), K from LDS activations, weights pre-swizzled in global.
// ---------------------------------------------------------------------------
__device__ __forceinline__ void gemm_acc(v8f acc[8][2],
    const __bf16* __restrict__ src, int ld,
    const __bf16* __restrict__ wt, int K,
    int lane, int wave) {
  const int kblocks = K >> 5;
  const int m  = lane & 15;
  const int hi = lane >> 4;
  for (int ks = 0; ks < kblocks; ++ks) {
    Frag b0, b1;
    {
      const __bf16* bp0 = wt + (((size_t)((wave * 2)     * kblocks + ks)) << 9) + (lane << 4);
      const __bf16* bp1 = wt + (((size_t)((wave * 2 + 1) * kblocks + ks)) << 9) + (lane << 4);
      b0.u[0] = *(const v4u*)bp0;  b0.u[1] = *(const v4u*)(bp0 + 8);
      b1.u[0] = *(const v4u*)bp1;  b1.u[1] = *(const v4u*)(bp1 + 8);
    }
    // A fragment: lanes 0-15 hold K+0..15, lanes 16-31 K+16..31
    const int ka = (ks << 5) + (hi << 4);
#pragma unroll
    for (int mt = 0; mt < 8; ++mt) {
      const __bf16* ap = src + (size_t)(mt * 16 + m) * ld + ka;
      Frag a;
      a.u[0] = *(const v4u*)ap;
      a.u[1] = *(const v4u*)(ap + 8);
      acc[mt][0] = __builtin_amdgcn_wmma_f32_16x16x32_bf16(
          false, a.v, false, b0.v, (short)0, acc[mt][0], false, false);
      acc[mt][1] = __builtin_amdgcn_wmma_f32_16x16x32_bf16(
          false, a.v, false, b1.v, (short)0, acc[mt][1], false, false);
    }
  }
}

__device__ __forceinline__ void zero_acc(v8f acc[8][2]) {
#pragma unroll
  for (int mt = 0; mt < 8; ++mt)
#pragma unroll
    for (int t = 0; t < 2; ++t)
#pragma unroll
      for (int i = 0; i < 8; ++i) acc[mt][t][i] = 0.0f;
}

// bias + ReLU + bf16 round + scatter to LDS (C layout: VGPR i -> M=i(+8), N=lane&15)
__device__ __forceinline__ void store_act(v8f acc[8][2], __bf16* dst, int ld,
                                          const float* __restrict__ bias,
                                          int lane, int wave) {
  const int m    = lane & 15;
  const int rofs = (lane >> 4) << 3;
#pragma unroll
  for (int t = 0; t < 2; ++t) {
    const int col = wave * 32 + t * 16 + m;
    const float bv = bias[col];
#pragma unroll
    for (int mt = 0; mt < 8; ++mt) {
#pragma unroll
      for (int i = 0; i < 8; ++i) {
        float v = acc[mt][t][i] + bv;
        v = v > 0.0f ? v : 0.0f;
        dst[(size_t)(mt * 16 + rofs + i) * ld + col] = f2bf(v);
      }
    }
  }
}

// ---------------------------------------------------------------------------
// Fused field kernel: encode -> 8-layer MLP (WMMA bf16) -> density head
// ---------------------------------------------------------------------------
__global__ __launch_bounds__(256, 1)
void mipnerf_field_kernel(const float* __restrict__ means,
                          const float* __restrict__ covs,
                          const float* __restrict__ basis,
                          const __bf16* __restrict__ W,
                          Biases bp,
                          const float* __restrict__ dens_w,
                          const float* __restrict__ dens_b,
                          float* __restrict__ out, int n_samples) {
  __shared__ __bf16 s_feat[MT * LDF];        // 133120 B
  __shared__ __bf16 s_act[2][MT * LDA];      // 135168 B   (total ~262 KB < 320 KB/WGP)

  const int tid   = threadIdx.x;
  const int tile0 = blockIdx.x * MT;

  // ---- encoding: one sample per thread (threads 0..127) ----
  if (tid < MT) {
    const int gi = tile0 + tid;
    const float x0 = means[gi * 3 + 0], x1 = means[gi * 3 + 1], x2 = means[gi * 3 + 2];
    float s = fmaxf(x0 * x0 + x1 * x1 + x2 * x2, 1e-32f);
    float g = 1.0f, tgp = 0.0f, c0 = x0, c1 = x1, c2 = x2;
    const bool inside = (s <= 1.0f);
    if (!inside) {
      float rs = sqrtf(s);
      g   = (2.0f * rs - 1.0f) / s;                   // (2*sqrt(s)-1)/s
      tgp = 2.0f * (-1.0f / (rs * s) + 1.0f / (s * s)); // 2*g'(s)
      c0 = g * x0; c1 = g * x1; c2 = g * x2;
    }
    const float* C = covs + (size_t)gi * 9;
    const float C00 = C[0], C01 = C[1], C02 = C[2], C11 = C[4], C12 = C[5], C22 = C[8];
    __bf16* fr = s_feat + (size_t)tid * LDF;
    for (int n = 0; n < NB; ++n) {
      const float b0v = basis[n], b1v = basis[NB + n], b2v = basis[2 * NB + n];
      const float xb = x0 * b0v + x1 * b1v + x2 * b2v;
      float fm, u0, u1, u2;
      if (inside) { fm = xb; u0 = b0v; u1 = b1v; u2 = b2v; }
      else {
        fm = g * xb;
        const float w = tgp * xb;                      // J = g*I + 2g' x x^T (symmetric)
        u0 = g * b0v + w * x0;
        u1 = g * b1v + w * x1;
        u2 = g * b2v + w * x2;
      }
      const float fv = C00 * u0 * u0 + C11 * u1 * u1 + C22 * u2 * u2
                     + 2.0f * (C01 * u0 * u1 + C02 * u0 * u2 + C12 * u1 * u2);
      float sc = 1.0f;
      for (int d = 0; d < NDEG; ++d) {
        const float sm = fm * sc;
        const float sv = fv * sc * sc;
        const float e  = __expf(-0.5f * sv);
        fr[d * NB + n]              = f2bf(e * __sinf(sm));  // sin(sm)
        fr[NDEG * NB + d * NB + n]  = f2bf(e * __cosf(sm));  // sin(sm + pi/2)
        sc *= 2.0f;
      }
    }
    for (int k = FEAT; k < LDF; ++k) fr[k] = f2bf(0.0f);     // K padding -> zero
  }
  __syncthreads();

  const int lane = tid & 31;
  const int wave = tid >> 5;
  v8f acc[8][2];

  // layer 0: feat(512 padded) -> act0
  zero_acc(acc);
  gemm_acc(acc, s_feat, LDF, W + OFF_W0, FEAT_PAD, lane, wave);
  store_act(acc, s_act[0], LDA, bp.b[0], lane, wave);
  __syncthreads();

  // layers 1..4 (ping-pong)
  for (int l = 1; l <= 4; ++l) {
    zero_acc(acc);
    gemm_acc(acc, s_act[(l - 1) & 1], LDA, W + OFF_W14 + (size_t)(l - 1) * 65536, NW, lane, wave);
    store_act(acc, s_act[l & 1], LDA, bp.b[l], lane, wave);
    __syncthreads();
  }

  // layer 5: concat(x, feat) == two accumulation passes
  zero_acc(acc);
  gemm_acc(acc, s_act[0], LDA, W + OFF_W5A, NW, lane, wave);
  gemm_acc(acc, s_feat,  LDF, W + OFF_W5B, FEAT_PAD, lane, wave);
  store_act(acc, s_act[1], LDA, bp.b[5], lane, wave);
  __syncthreads();

  // layer 6
  zero_acc(acc);
  gemm_acc(acc, s_act[1], LDA, W + OFF_W6, NW, lane, wave);
  store_act(acc, s_act[0], LDA, bp.b[6], lane, wave);
  __syncthreads();

  // layer 7
  zero_acc(acc);
  gemm_acc(acc, s_act[0], LDA, W + OFF_W7, NW, lane, wave);
  store_act(acc, s_act[1], LDA, bp.b[7], lane, wave);
  __syncthreads();

  // ---- density head: softplus(x @ dens_w + dens_b - 1) ----
  if (tid < MT) {
    const __bf16* xr = s_act[1] + (size_t)tid * LDA;
    float sum = 0.0f;
    for (int c = 0; c < NW; ++c) sum += bf2f(xr[c]) * dens_w[c];
    const float z  = sum + dens_b[0] - 1.0f;
    const float sp = fmaxf(z, 0.0f) + log1pf(__expf(-fabsf(z)));
    out[tile0 + tid] = sp;
  }
  // ---- write final activations x (fp32) ----
  float* outx = out + n_samples;
  for (int idx = tid; idx < MT * NW; idx += 256) {
    const int r = idx >> 8;
    const int c = idx & 255;
    outx[(size_t)(tile0 + r) * NW + c] = bf2f(s_act[1][(size_t)r * LDA + c]);
  }
}

// ---------------------------------------------------------------------------
// Host launch
// ---------------------------------------------------------------------------
extern "C" void kernel_launch(void* const* d_in, const int* in_sizes, int n_in,
                              void* d_out, int out_size, void* d_ws, size_t ws_size,
                              hipStream_t stream) {
  const float* means = (const float*)d_in[0];
  const float* covs  = (const float*)d_in[1];
  const float* basis = (const float*)d_in[2];
  const float* pw[8];
  Biases bp;
  for (int i = 0; i < 8; ++i) {
    pw[i]   = (const float*)d_in[3 + i];
    bp.b[i] = (const float*)d_in[11 + i];
  }
  const float* dens_w = (const float*)d_in[19];
  const float* dens_b = (const float*)d_in[20];
  __bf16* W = (__bf16*)d_ws;   // needs ~1.44 MB of workspace

  auto prep = [&](const float* src, __bf16* dst, int Ksrc, int Kpad, int rowOff) {
    int total = NW * Kpad;
    prep_wt<<<(total + 255) / 256, 256, 0, stream>>>(src, dst, Ksrc, Kpad, rowOff);
  };
  prep(pw[0], W + OFF_W0, FEAT, FEAT_PAD, 0);
  for (int l = 1; l <= 4; ++l) prep(pw[l], W + OFF_W14 + (size_t)(l - 1) * 65536, NW, NW, 0);
  prep(pw[5], W + OFF_W5A, NW, NW, 0);
  prep(pw[5], W + OFF_W5B, FEAT, FEAT_PAD, NW);   // rows 256..759 of the 760-row skip matrix
  prep(pw[6], W + OFF_W6, NW, NW, 0);
  prep(pw[7], W + OFF_W7, NW, NW, 0);

  const int n_samples = in_sizes[0] / 3;          // 131072
  mipnerf_field_kernel<<<n_samples / MT, 256, 0, stream>>>(
      means, covs, basis, W, bp, dens_w, dens_b, (float*)d_out, n_samples);
}
